// MultiHeadSelfAttention_64115271795316
// MI455X (gfx1250) — matrix operand order repaired and checked by reading, not compiled, not run on hardware
//
#include <hip/hip_runtime.h>
#include <hip/hip_bf16.h>

// CDNA5 / gfx1250 multi-head attention: bf16 WMMA + Tensor Data Mover.
// B=4, S=2048, D=1024, H=16, Dh=64.
// All global->LDS staging is done by TDM (tensor_load_to_lds), double-buffered
// and overlapped with WMMA compute via TENSORcnt waits. No VGPR staging.

typedef __attribute__((ext_vector_type(16))) __bf16 v16bf;
typedef __attribute__((ext_vector_type(8)))  float  v8f;
typedef __attribute__((ext_vector_type(4)))  unsigned int v4u;
typedef __attribute__((ext_vector_type(8)))  int v8i_t;
typedef __attribute__((ext_vector_type(4)))  int v4i_t;

#define WMMA_BF16(a, b, c) \
  __builtin_amdgcn_wmma_f32_16x16x32_bf16(false, (a), false, (b), (short)0, (c), false, false)

union Frag16 {
  v16bf  v;
  uint4  q[2];
  __bf16 h[16];
};

// ---------------------------------------------------------------------------
// TDM 2D tile load: global (bf16, row-major, row pitch stride0 elems) -> LDS.
// Pads LDS destination by pad DWORDs every interval DWORDs (ISA D# group1).
// Issued once per wave that executes it; tracked by TENSORcnt.
// ---------------------------------------------------------------------------
__device__ __forceinline__ void tdm_load_2d(
    unsigned lds_off, const void* gptr,
    unsigned tensor_d0, unsigned tensor_d1,     // full tensor extents (elems)
    unsigned tile_d0, unsigned tile_d1,         // tile extents (elems)
    unsigned long long stride0,                 // row pitch (elems)
    unsigned pad_interval_code, unsigned pad_amount_code) {
  unsigned long long ga = (unsigned long long)(uintptr_t)gptr;
  v4u g0;
  g0[0] = 1u;                                    // count=1 (valid), user mode
  g0[1] = lds_off;                               // lds_addr (bytes)
  g0[2] = (unsigned)(ga & 0xFFFFFFFFu);          // global_addr[31:0]
  g0[3] = (unsigned)((ga >> 32) & 0x1FFFFFFu)    // global_addr[56:32]
        | (2u << 30);                            // type=2 ("image")
  v8i_t g1;
  g1[0] = (int)((1u << 16)                       // data_size=1 -> 2 bytes
              | (1u << 20)                       // pad_enable
              | (pad_interval_code << 22)
              | (pad_amount_code << 25));
  g1[1] = (int)((tensor_d0 & 0xFFFFu) << 16);            // tensor_dim0 lo16
  g1[2] = (int)(((tensor_d0 >> 16) & 0xFFFFu)
              | ((tensor_d1 & 0xFFFFu) << 16));          // d0 hi16 | d1 lo16
  g1[3] = (int)(((tensor_d1 >> 16) & 0xFFFFu)
              | ((tile_d0 & 0xFFFFu) << 16));            // d1 hi16 | tile0
  g1[4] = (int)(tile_d1 & 0xFFFFu);                      // tile1 | tile2=0
  g1[5] = (int)(unsigned)(stride0 & 0xFFFFFFFFu);        // dim0_stride lo32
  g1[6] = (int)(unsigned)((stride0 >> 32) & 0xFFFFu);    // dim0_stride hi16
  g1[7] = 0;
  v4i_t z4 = {0, 0, 0, 0};
  v8i_t z8i = {0, 0, 0, 0, 0, 0, 0, 0};
  // 6-arg form (this toolchain): groups 2/3 + trailing group unused for 2D.
  __builtin_amdgcn_tensor_load_to_lds(g0, g1, z4, z4, z8i, 0);
}

__device__ __forceinline__ unsigned lds_off_of(const void* p) {
  return (unsigned)(uintptr_t)p;   // flat LDS address low 32 bits = LDS offset
}

// pad codes for 64-elem bf16 rows -> pitch 72 elems:
// row = 128B = 32 DWORDs -> interval code 4 (2^(c+1)=32); pad 16B = 4 DWORDs
// -> amount code 3 (c+1=4).
#define PAD_I 4u
#define PAD_A 3u

// ---------------------------------------------------------------------------
// fp32 -> bf16 conversion (grid-stride)
// ---------------------------------------------------------------------------
__global__ void cvt_f32_bf16_kernel(const float* __restrict__ in,
                                    __bf16* __restrict__ out, int n) {
  int i = blockIdx.x * blockDim.x + threadIdx.x;
  int stride = gridDim.x * blockDim.x;
  for (; i < n; i += stride) out[i] = (__bf16)in[i];
}

// ---------------------------------------------------------------------------
// fp32 -> bf16 transpose-convert: out[n*K + k] = (bf16) in[k*N + n]
// ---------------------------------------------------------------------------
__global__ __launch_bounds__(256) void cvtT_f32_bf16_kernel(
    const float* __restrict__ in, __bf16* __restrict__ out, int K, int N) {
  __shared__ float tile[32][33];
  const int k0 = blockIdx.y * 32, n0 = blockIdx.x * 32;
  const int tx = threadIdx.x & 31, ty = threadIdx.x >> 5;
#pragma unroll
  for (int i = 0; i < 4; ++i) {
    int r = ty + i * 8;
    tile[r][tx] = in[(size_t)(k0 + r) * N + n0 + tx];
  }
  __syncthreads();
#pragma unroll
  for (int i = 0; i < 4; ++i) {
    int r = ty + i * 8;
    out[(size_t)(n0 + r) * K + k0 + tx] = (__bf16)tile[tx][r];
  }
}

// ---------------------------------------------------------------------------
// bf16 WMMA GEMM: C[M,N] = A[M,K] * Bt[N,K]^T + bias[N]
// Block tile 128x128x64, 256 threads (8 waves, 2x4), each wave 64x32.
// TDM double-buffered staging: wave0 issues tensor loads for tile k+1,
// s_wait_tensorcnt gates tile k; DMA overlaps the 16-WMMA compute.
// mode 0: write fp32 to Cout
// mode 1: scatter bf16 Q/K ([B,H,S,Dh]) and V^T ([B,H,Dh,S])
// ---------------------------------------------------------------------------
__global__ __launch_bounds__(256) void gemm_bf16_kernel(
    const __bf16* __restrict__ A, const __bf16* __restrict__ Bt,
    const float* __restrict__ bias, int M, int N, int K, int mode,
    float* __restrict__ Cout, __bf16* __restrict__ Qo,
    __bf16* __restrict__ Ko, __bf16* __restrict__ Vo) {
  constexpr int BM = 128, BN = 128, BK = 64, PAD = 8;
  __shared__ __bf16 lA[2][BM][BK + PAD];
  __shared__ __bf16 lB[2][BN][BK + PAD];

  const int tid  = threadIdx.x;
  const int lane = tid & 31;
  const int wave = tid >> 5;
  const int wm = wave & 1, wn = wave >> 1;  // wave grid 2(M) x 4(N)
  const int lr = lane & 15, hi = lane >> 4;
  const int mbase = blockIdx.y * BM;
  const int nbase = blockIdx.x * BN;

  const v8f z8 = {0, 0, 0, 0, 0, 0, 0, 0};
  v8f acc[4][2];
#pragma unroll
  for (int i = 0; i < 4; ++i)
#pragma unroll
    for (int j = 0; j < 2; ++j) acc[i][j] = z8;

  const __bf16* ag = A  + (size_t)mbase * K;
  const __bf16* bg = Bt + (size_t)nbase * K;

  // prologue: DMA tile 0 into buffer 0
  if (wave == 0) {
    tdm_load_2d(lds_off_of(&lA[0][0][0]), ag, K, M, BK, BM, K, PAD_I, PAD_A);
    tdm_load_2d(lds_off_of(&lB[0][0][0]), bg, K, N, BK, BN, K, PAD_I, PAD_A);
  }

  int buf = 0;
  for (int k0 = 0; k0 < K; k0 += BK) {
    const bool hasNext = (k0 + BK < K);
    if (wave == 0) {
      if (hasNext) {
        tdm_load_2d(lds_off_of(&lA[buf ^ 1][0][0]), ag + k0 + BK,
                    K, M, BK, BM, K, PAD_I, PAD_A);
        tdm_load_2d(lds_off_of(&lB[buf ^ 1][0][0]), bg + k0 + BK,
                    K, N, BK, BN, K, PAD_I, PAD_A);
        // TDM completes in order: <=2 outstanding means tile k0 has landed.
        __builtin_amdgcn_s_wait_tensorcnt(2);
      } else {
        __builtin_amdgcn_s_wait_tensorcnt(0);
      }
    }
    __syncthreads();

    // ---- 2 K-steps x 8 WMMAs from LDS tile `buf` ----
#pragma unroll
    for (int ks = 0; ks < 2; ++ks) {
      const int kb = ks * 32;
      Frag16 fa[4], fb[2];
      const int c0a = kb + (hi ? 8 : 0);
#pragma unroll
      for (int mt = 0; mt < 4; ++mt) {
        int r = wm * 64 + mt * 16 + lr;
        fa[mt].q[0] = *(const uint4*)&lA[buf][r][c0a];
        fa[mt].q[1] = *(const uint4*)&lA[buf][r][c0a + 16];
      }
#pragma unroll
      for (int nt = 0; nt < 2; ++nt) {
        int n = wn * 32 + nt * 16 + lr;
        fb[nt].q[0] = *(const uint4*)&lB[buf][n][kb + hi * 8];
        fb[nt].q[1] = *(const uint4*)&lB[buf][n][kb + hi * 8 + 16];
      }
#pragma unroll
      for (int mt = 0; mt < 4; ++mt)
#pragma unroll
        for (int nt = 0; nt < 2; ++nt)
          acc[mt][nt] = WMMA_BF16(fa[mt].v, fb[nt].v, acc[mt][nt]);
    }
    __syncthreads();   // all waves done reading `buf` before it is re-DMAed
    buf ^= 1;
  }

  // ---- epilogue: C layout VGPR r -> row (r + 8*hi), col = lr ----
#pragma unroll
  for (int mt = 0; mt < 4; ++mt) {
#pragma unroll
    for (int nt = 0; nt < 2; ++nt) {
#pragma unroll
      for (int r = 0; r < 8; ++r) {
        int grow = mbase + wm * 64 + mt * 16 + r + hi * 8;
        int gcol = nbase + wn * 32 + nt * 16 + lr;
        float val = acc[mt][nt][r] + bias[gcol];
        if (mode == 0) {
          Cout[(size_t)grow * N + gcol] = val;
        } else {
          int which = gcol >> 10;
          int d = gcol & 1023;
          int hh = d >> 6, dh = d & 63;
          int bb = grow >> 11, ss = grow & 2047;
          size_t bhid = (size_t)bb * 16 + hh;
          __bf16 bv = (__bf16)val;
          if (which == 0)      Qo[(bhid * 2048 + ss) * 64 + dh] = bv;
          else if (which == 1) Ko[(bhid * 2048 + ss) * 64 + dh] = bv;
          else                 Vo[(bhid * 64 + dh) * 2048 + ss] = bv;  // V^T
        }
      }
    }
  }
}

// ---------------------------------------------------------------------------
// Flash attention: one block = (b,h) x 64-query tile; 4 waves x 16 rows.
// 64-key K / V^T tiles DMAed by TDM (double-buffered), 16 WMMAs per tile.
// ---------------------------------------------------------------------------
__global__ __launch_bounds__(128) void attn_kernel(
    const __bf16* __restrict__ Q, const __bf16* __restrict__ Kmat,
    const __bf16* __restrict__ VT, __bf16* __restrict__ O) {
  constexpr int S = 2048, Dh = 64, KT = 64, PAD = 8;
  __shared__ __bf16 lK[2][KT][Dh + PAD];     // [key][dh]
  __shared__ __bf16 lVT[2][Dh][KT + PAD];    // [dh][key]
  __shared__ __bf16 lP[4][16][KT + PAD];     // per-wave P tile [row][key]

  const int tid  = threadIdx.x;
  const int lane = tid & 31, w = tid >> 5;
  const int lr = lane & 15, hi = lane >> 4;
  const int bh = blockIdx.x;                 // b*16 + h
  const int b = bh >> 4, h = bh & 15;
  const int qbase = blockIdx.y * 64 + w * 16;

  const __bf16* Qh  = Q    + (size_t)bh * S * Dh;
  const __bf16* Kh  = Kmat + (size_t)bh * S * Dh;
  const __bf16* VTh = VT   + (size_t)bh * Dh * S;

  // prologue: DMA KV tile 0 into buffer 0
  if (w == 0) {
    tdm_load_2d(lds_off_of(&lK[0][0][0]), Kh, Dh, S, Dh, KT, Dh, PAD_I, PAD_A);
    tdm_load_2d(lds_off_of(&lVT[0][0][0]), VTh, S, Dh, KT, Dh, S, PAD_I, PAD_A);
  }

  // Q fragments: A-layout over two Dh chunks (K=0..31, K=32..63)
  Frag16 aLo, aHi;
  {
    const __bf16* qp = Qh + (size_t)(qbase + lr) * Dh;
    const int c0 = hi ? 8 : 0;
    aLo.q[0] = *(const uint4*)(qp + c0);
    aLo.q[1] = *(const uint4*)(qp + c0 + 16);
    aHi.q[0] = *(const uint4*)(qp + 32 + c0);
    aHi.q[1] = *(const uint4*)(qp + 32 + c0 + 16);
  }

  const v8f z8 = {0, 0, 0, 0, 0, 0, 0, 0};
  v8f o[4];
#pragma unroll
  for (int t = 0; t < 4; ++t) o[t] = z8;
  float m[8], l[8];
#pragma unroll
  for (int r = 0; r < 8; ++r) { m[r] = -3.0e38f; l[r] = 0.0f; }
  const float scale = 0.125f;  // 1/sqrt(Dh)

  int buf = 0;
  for (int j = 0; j < S; j += KT) {
    const bool hasNext = (j + KT < S);
    if (w == 0) {
      if (hasNext) {
        tdm_load_2d(lds_off_of(&lK[buf ^ 1][0][0]), Kh + (size_t)(j + KT) * Dh,
                    Dh, S, Dh, KT, Dh, PAD_I, PAD_A);
        tdm_load_2d(lds_off_of(&lVT[buf ^ 1][0][0]), VTh + j + KT,
                    S, Dh, KT, Dh, S, PAD_I, PAD_A);
        __builtin_amdgcn_s_wait_tensorcnt(2);
      } else {
        __builtin_amdgcn_s_wait_tensorcnt(0);
      }
    }
    __syncthreads();

    // ---- scores: Q (16xDh) * K^T (Dh x 64) -> four 16x16 tiles ----
    v8f c[4];
#pragma unroll
    for (int n = 0; n < 4; ++n) {
      Frag16 bLo, bHi;
      const __bf16* kp0 = &lK[buf][n * 16 + lr][hi * 8];
      bLo.q[0] = *(const uint4*)kp0;  bLo.q[1] = *(const uint4*)(kp0 + 16);
      const __bf16* kp1 = &lK[buf][n * 16 + lr][32 + hi * 8];
      bHi.q[0] = *(const uint4*)kp1;  bHi.q[1] = *(const uint4*)(kp1 + 16);
      c[n] = z8;
      c[n] = WMMA_BF16(aLo.v, bLo.v, c[n]);
      c[n] = WMMA_BF16(aHi.v, bHi.v, c[n]);
    }

    // ---- online softmax; row = (r + 8*hi), reductions over 16 lanes ----
#pragma unroll
    for (int r = 0; r < 8; ++r) {
      float s0 = c[0][r] * scale, s1 = c[1][r] * scale;
      float s2 = c[2][r] * scale, s3 = c[3][r] * scale;
      float mx = fmaxf(fmaxf(s0, s1), fmaxf(s2, s3));
      mx = fmaxf(mx, __shfl_xor(mx, 1, 32));
      mx = fmaxf(mx, __shfl_xor(mx, 2, 32));
      mx = fmaxf(mx, __shfl_xor(mx, 4, 32));
      mx = fmaxf(mx, __shfl_xor(mx, 8, 32));
      float mn = fmaxf(m[r], mx);
      float p0 = __expf(s0 - mn), p1 = __expf(s1 - mn);
      float p2 = __expf(s2 - mn), p3 = __expf(s3 - mn);
      float rs = (p0 + p1) + (p2 + p3);
      rs += __shfl_xor(rs, 1, 32);
      rs += __shfl_xor(rs, 2, 32);
      rs += __shfl_xor(rs, 4, 32);
      rs += __shfl_xor(rs, 8, 32);
      float alpha = __expf(m[r] - mn);
      l[r] = l[r] * alpha + rs;
      m[r] = mn;
#pragma unroll
      for (int t = 0; t < 4; ++t) o[t][r] *= alpha;
      int row = r + hi * 8;
      lP[w][row][lr]      = (__bf16)p0;
      lP[w][row][16 + lr] = (__bf16)p1;
      lP[w][row][32 + lr] = (__bf16)p2;
      lP[w][row][48 + lr] = (__bf16)p3;
    }
    __syncthreads();

    // ---- P (16x64) * V (64xDh): 2 key-chunks x 4 Dh slices ----
    {
      Frag16 pf[2];
      const int c0a = hi ? 8 : 0;
#pragma unroll
      for (int cc = 0; cc < 2; ++cc) {
        pf[cc].q[0] = *(const uint4*)&lP[w][lr][cc * 32 + c0a];
        pf[cc].q[1] = *(const uint4*)&lP[w][lr][cc * 32 + c0a + 16];
      }
#pragma unroll
      for (int t = 0; t < 4; ++t) {
#pragma unroll
        for (int cc = 0; cc < 2; ++cc) {
          Frag16 vf;
          const __bf16* vp = &lVT[buf][t * 16 + lr][cc * 32 + hi * 8];
          vf.q[0] = *(const uint4*)vp;
          vf.q[1] = *(const uint4*)(vp + 16);
          o[t] = WMMA_BF16(pf[cc].v, vf.v, o[t]);
        }
      }
    }
    __syncthreads();   // everyone done with `buf` before it is re-DMAed
    buf ^= 1;
  }

  // ---- normalize and store in combined-head layout [B,S,D] (bf16) ----
#pragma unroll
  for (int r = 0; r < 8; ++r) {
    float inv = 1.0f / l[r];
    int srowg = qbase + r + hi * 8;
#pragma unroll
    for (int t = 0; t < 4; ++t) {
      size_t dst = ((size_t)b * S + srowg) * 1024 + h * 64 + t * 16 + lr;
      O[dst] = (__bf16)(o[t][r] * inv);
    }
  }
}

// ---------------------------------------------------------------------------
// Launch
// ---------------------------------------------------------------------------
extern "C" void kernel_launch(void* const* d_in, const int* in_sizes, int n_in,
                              void* d_out, int out_size, void* d_ws,
                              size_t ws_size, hipStream_t stream) {
  const int B = 4, S = 2048, D = 1024, T = B * S;  // 8192 tokens
  const float* x     = (const float*)d_in[0];
  const float* w_qkv = (const float*)d_in[1];
  const float* b_qkv = (const float*)d_in[2];
  const float* w_out = (const float*)d_in[3];
  const float* b_out = (const float*)d_in[4];
  float* out = (float*)d_out;

  // workspace carve-up (bf16), total 88 MB
  __bf16* xb     = (__bf16*)d_ws;                  // [T, D]
  __bf16* wqkvT  = xb    + (size_t)T * D;          // [3D, D]  (transposed)
  __bf16* woutT  = wqkvT + (size_t)D * 3 * D;      // [D, D]   (transposed)
  __bf16* Qb     = woutT + (size_t)D * D;          // [B,H,S,Dh]
  __bf16* Kb     = Qb    + (size_t)T * D;          // [B,H,S,Dh]
  __bf16* VTb    = Kb    + (size_t)T * D;          // [B,H,Dh,S]
  __bf16* Ob     = VTb   + (size_t)T * D;          // [B,S,D]

  cvt_f32_bf16_kernel<<<2048, 256, 0, stream>>>(x, xb, T * D);
  cvtT_f32_bf16_kernel<<<dim3(3 * D / 32, D / 32), 256, 0, stream>>>(
      w_qkv, wqkvT, D, 3 * D);
  cvtT_f32_bf16_kernel<<<dim3(D / 32, D / 32), 256, 0, stream>>>(
      w_out, woutT, D, D);

  // QKV projection: [8192,1024] x [1024,3072], scatter to Q/K/V^T head layout
  gemm_bf16_kernel<<<dim3(3 * D / 128, T / 128), 256, 0, stream>>>(
      xb, wqkvT, b_qkv, T, 3 * D, D, /*mode=*/1, nullptr, Qb, Kb, VTb);

  // flash attention: 64 (b,h) pairs x 32 query tiles
  attn_kernel<<<dim3(64, S / 64), 128, 0, stream>>>(Qb, Kb, VTb, Ob);

  // output projection: [8192,1024] x [1024,1024] -> fp32 d_out
  gemm_bf16_kernel<<<dim3(D / 128, T / 128), 256, 0, stream>>>(
      Ob, woutT, b_out, T, D, D, /*mode=*/0, out, nullptr, nullptr, nullptr);
}
